// EfficientNodeLabelling_22368189678010
// MI455X (gfx1250) — matrix operand order repaired and checked by reading, not compiled, not run on hardware
//
#include <hip/hip_runtime.h>

// ---------------------------------------------------------------------------
// EfficientNodeLabelling forward for MI455X (gfx1250, wave32, WMMA)
//
// Phase 1: bit-packed set algebra for the DE+ counts (replaces dense A@A).
// Phase 2: bf16 WMMA GEMMs with double-buffered async global->LDS staging.
// ---------------------------------------------------------------------------

typedef __attribute__((ext_vector_type(16))) __bf16       v16bf;
typedef __attribute__((ext_vector_type(8)))  float        v8f;
typedef __attribute__((ext_vector_type(8)))  unsigned int v8u;

#define N_NODES 4096
#define NWORDS  128            // 4096 bits / 32
#define E_EDGES 8192
#define C_IN    256
#define HID     256
#define KPAD    288            // 261 (C+5) padded up to 9 * 32

// ---------------------------------------------------------------------------
// Kernel 1: pack adjacency (bool bytes) into bitset rows (128 u32 per node)
// ---------------------------------------------------------------------------
__global__ void k_pack_bits(const unsigned char* __restrict__ adj,
                            unsigned int* __restrict__ Ap) {
  int w = blockIdx.x * blockDim.x + threadIdx.x;     // global word id
  if (w >= N_NODES * NWORDS) return;
  int row = w >> 7;
  int wc  = w & 127;
  const unsigned char* p = adj + (long)row * N_NODES + wc * 32;
  unsigned int bits = 0u;
#pragma unroll
  for (int b = 0; b < 32; ++b) bits |= (p[b] ? 1u : 0u) << b;
  Ap[w] = bits;
}

// ---------------------------------------------------------------------------
// Kernel 2: strict 2-hop rows: A2[i] = (union of A[k] for k in N(i)) & ~A[i],
// diagonal cleared.  One 128-thread block per node; row i staged in LDS.
// ---------------------------------------------------------------------------
__global__ void __launch_bounds__(128)
k_twohop(const unsigned int* __restrict__ Ap, unsigned int* __restrict__ A2p) {
  __shared__ unsigned int rowi[NWORDS];
  int i = blockIdx.x;
  int t = threadIdx.x;                               // word index 0..127
  rowi[t] = Ap[(long)i * NWORDS + t];
  __syncthreads();
  unsigned int acc = 0u;
  for (int wk = 0; wk < NWORDS; ++wk) {
    unsigned int bits = rowi[wk];
    while (bits) {
      int k = (wk << 5) + __ffs(bits) - 1;           // neighbor id
      bits &= bits - 1u;
      acc |= Ap[(long)k * NWORDS + t];
    }
  }
  acc &= ~rowi[t];                                   // not adjacent
  if (t == (i >> 5)) acc &= ~(1u << (i & 31));       // not self
  A2p[(long)i * NWORDS + t] = acc;
}

// ---------------------------------------------------------------------------
// Kernel 3: one wave32 per edge.
//   - 5 DE+ counts via popcount over 128-word bitsets
//   - struct_encode (5x5 linear + BN-eval + relu)
//   - fused feature build: feat[e] = [bf16(x[u]*x[v]) (256) | h (5) | 0 pad]
// ---------------------------------------------------------------------------
__global__ void __launch_bounds__(256)
k_edge_feat(const unsigned int* __restrict__ Ap,
            const unsigned int* __restrict__ A2p,
            const int* __restrict__ edges,
            const float* __restrict__ x,
            const float* __restrict__ Ws, const float* __restrict__ bs,
            const float* __restrict__ gamma, const float* __restrict__ beta,
            __bf16* __restrict__ feat) {
  int lane = threadIdx.x & 31;
  int e = blockIdx.x * (blockDim.x >> 5) + (threadIdx.x >> 5);
  if (e >= E_EDGES) return;
  int u = edges[e];
  int v = edges[E_EDGES + e];

  unsigned int c11 = 0, c12 = 0, c1i = 0, c22 = 0, c2i = 0;
#pragma unroll
  for (int j = 0; j < 4; ++j) {
    int w = lane + (j << 5);                         // word 0..127
    unsigned int mask = 0xffffffffu;
    if (w == (u >> 5)) mask &= ~(1u << (u & 31));    // exclude endpoint u
    if (w == (v >> 5)) mask &= ~(1u << (v & 31));    // exclude endpoint v
    unsigned int d1u = Ap [(long)u * NWORDS + w] & mask;
    unsigned int d1v = Ap [(long)v * NWORDS + w] & mask;
    unsigned int d2u = A2p[(long)u * NWORDS + w] & mask;
    unsigned int d2v = A2p[(long)v * NWORDS + w] & mask;
    unsigned int diu = mask & ~d1u & ~d2u;
    unsigned int div_ = mask & ~d1v & ~d2v;
    c11 += __popc(d1u & d1v);
    c12 += __popc(d1u & d2v) + __popc(d2u & d1v);
    c1i += __popc(d1u & div_) + __popc(diu & d1v);
    c22 += __popc(d2u & d2v);
    c2i += __popc(d2u & div_) + __popc(diu & d2v);
  }
  // wave32 reduction
  for (int off = 16; off > 0; off >>= 1) {
    c11 += __shfl_down((int)c11, off);
    c12 += __shfl_down((int)c12, off);
    c1i += __shfl_down((int)c1i, off);
    c22 += __shfl_down((int)c22, off);
    c2i += __shfl_down((int)c2i, off);
  }
  float s0 = (float)__shfl((int)c11, 0);
  float s1 = (float)__shfl((int)c12, 0);
  float s2 = (float)__shfl((int)c1i, 0);
  float s3 = (float)__shfl((int)c22, 0);
  float s4 = (float)__shfl((int)c2i, 0);

  // fused feature row (bf16, K padded to 288)
  const float* xu = x + (long)u * C_IN;
  const float* xv = x + (long)v * C_IN;
  __bf16* fr = feat + (long)e * KPAD;
#pragma unroll
  for (int j = 0; j < 8; ++j) {
    int c = lane * 8 + j;
    fr[c] = (__bf16)(xu[c] * xv[c]);
  }
  if (lane < 5) {
    float h = bs[lane];
    h += s0 * Ws[0 * 5 + lane];
    h += s1 * Ws[1 * 5 + lane];
    h += s2 * Ws[2 * 5 + lane];
    h += s3 * Ws[3 * 5 + lane];
    h += s4 * Ws[4 * 5 + lane];
    h = h * (gamma[lane] * rsqrtf(1.0f + 1e-5f)) + beta[lane];
    fr[C_IN + lane] = (__bf16)fmaxf(h, 0.0f);
  } else {
    fr[C_IN + lane] = (__bf16)0.0f;                  // zero pad cols 261..287
  }
}

// ---------------------------------------------------------------------------
// Kernel 4: weight conversion with TRANSPOSE (done once, reused by 128 blocks)
//   W1 (261x256 f32) -> W1t (256 x 288 bf16, [n][k], zero-padded k)
//   W2 (256x256 f32) -> W2t (256 x 256 bf16, [n][k])
// ---------------------------------------------------------------------------
__global__ void k_cvt_weights(const float* __restrict__ W1,
                              const float* __restrict__ W2,
                              __bf16* __restrict__ W1t,
                              __bf16* __restrict__ W2t) {
  int tid = blockIdx.x * blockDim.x + threadIdx.x;
  const int n1 = HID * KPAD;                         // 73728
  if (tid < n1) {
    int n = tid / KPAD, k = tid % KPAD;
    W1t[tid] = (__bf16)((k < 261) ? W1[k * 256 + n] : 0.0f);
  } else if (tid < n1 + HID * HID) {
    int t = tid - n1;
    int n = t >> 8, k = t & 255;
    W2t[t] = (__bf16)W2[k * 256 + n];
  }
}

// ---------------------------------------------------------------------------
// Kernel 5: bf16 WMMA GEMM:  C = relu(A[M,K] @ B + bias),  B given as BT[N][K].
// 256 threads = 8 wave32; 64x64 tile/block; each wave: one A frag reused for
// two B frags -> 2x v_wmma_f32_16x16x32_bf16 per K-step.
// Staging: double-buffered LDS fed by GLOBAL_LOAD_ASYNC_TO_LDS_B128
// (ASYNCcnt-tracked, no VGPR transit); steady-state loop keeps the next
// k-tile in flight; last iteration peeled for straight-line control flow.
// ---------------------------------------------------------------------------
__global__ void __launch_bounds__(256)
k_gemm_relu_wmma(const __bf16* __restrict__ A, const __bf16* __restrict__ BT,
                 const float* __restrict__ bias, __bf16* __restrict__ Cout,
                 int M, int Nn, int K) {
  __shared__ __bf16 As[2 * 64 * 32];                 // double-buffered A tile
  __shared__ __bf16 Bs[2 * 64 * 32];                 // double-buffered B tile [n][k]
  const int tid  = threadIdx.x;
  const int lane = tid & 31;
  const int wave = tid >> 5;
  const int m0 = blockIdx.y * 64;
  const int n0 = blockIdx.x * 64;
  const int wm = wave & 3;                           // 0..3 -> 16-row strip
  const int wn = wave >> 2;                          // 0..1 -> 32-col strip

  // per-thread staging slot: 256 threads x 16B = one 64x32 bf16 tile
  const int rs = tid >> 2;                           // tile row 0..63
  const int cs = (tid & 3) * 8;                      // tile col 0,8,16,24
  const __bf16* Ag = A  + (long)(m0 + rs) * K + cs;
  const __bf16* Bg = BT + (long)(n0 + rs) * K + cs;
  __bf16* Al = As + rs * 32 + cs;
  __bf16* Bl = Bs + rs * 32 + cs;

  v8f acc0 = {};
  v8f acc1 = {};
  const unsigned int* As32 = (const unsigned int*)As;
  const unsigned int* Bt32 = (const unsigned int*)Bs;
  const int nk = K >> 5;

#if defined(__gfx1250__)
  // async global->LDS copy of both 16B slots for k-tile t into buffer t&1
  #define ISSUE_TILE(t)                                                         \
    { unsigned la = (unsigned)(uintptr_t)(Al + ((t) & 1) * 2048);               \
      unsigned lb = (unsigned)(uintptr_t)(Bl + ((t) & 1) * 2048);               \
      unsigned long long ga = (unsigned long long)(uintptr_t)(Ag + (t) * 32);   \
      unsigned long long gb = (unsigned long long)(uintptr_t)(Bg + (t) * 32);   \
      asm volatile("global_load_async_to_lds_b128 %0, %1, off\n\t"              \
                   "global_load_async_to_lds_b128 %2, %3, off"                  \
                   :: "v"(la), "v"(ga), "v"(lb), "v"(gb) : "memory"); }
  #define WAIT_NEXT() asm volatile("s_wait_asynccnt 0x2" ::: "memory")
  #define WAIT_ALL()  asm volatile("s_wait_asynccnt 0x0" ::: "memory")
#else
  #define ISSUE_TILE(t)                                                         \
    { *(uint4*)(Al + ((t) & 1) * 2048) = *(const uint4*)(Ag + (t) * 32);        \
      *(uint4*)(Bl + ((t) & 1) * 2048) = *(const uint4*)(Bg + (t) * 32); }
  #define WAIT_NEXT()
  #define WAIT_ALL()
#endif

  // one k-tile of compute: A frag + two B frags -> 2 WMMA (all b128 LDS reads)
  auto compute_tile = [&](int t) {
    const unsigned int* Ab = As32 + (t & 1) * 1024;  // 2048 bf16 = 1024 u32
    const unsigned int* Bb = Bt32 + (t & 1) * 1024;
    // A fragment: 16x32 bf16, ISA lane layout; two aligned 16B LDS reads
    int row = wm * 16 + (lane & 15);
    int kba = (lane < 16) ? 0 : 4;                   // dword offset 0 or 4
    const uint4* pa = (const uint4*)(Ab + row * 16 + kba);
    uint4 a0 = pa[0];                                // K = kb..kb+7
    uint4 a1 = pa[2];                                // K = 16+kb..16+kb+7
    v8u ar;
    ar[0] = a0.x; ar[1] = a0.y; ar[2] = a0.z; ar[3] = a0.w;
    ar[4] = a1.x; ar[5] = a1.y; ar[6] = a1.z; ar[7] = a1.w;
    v16bf af = __builtin_bit_cast(v16bf, ar);
#pragma unroll
    for (int sub = 0; sub < 2; ++sub) {
      int n   = wn * 32 + sub * 16 + (lane & 15);
      int kbb = (lane < 16) ? 0 : 4;                 // dword offset 0 or 4
      const uint4* pb = (const uint4*)(Bb + n * 16 + kbb);
      uint4 b0 = pb[0];
      uint4 b1 = pb[2];
      v8u br;
      br[0] = b0.x; br[1] = b0.y; br[2] = b0.z; br[3] = b0.w;
      br[4] = b1.x; br[5] = b1.y; br[6] = b1.z; br[7] = b1.w;
      v16bf bf = __builtin_bit_cast(v16bf, br);
#if defined(__gfx1250__)
      if (sub == 0)
        acc0 = __builtin_amdgcn_wmma_f32_16x16x32_bf16(
            false, af, false, bf, (short)0, acc0, false, false);
      else
        acc1 = __builtin_amdgcn_wmma_f32_16x16x32_bf16(
            false, af, false, bf, (short)0, acc1, false, false);
#else
      (void)bf; (void)af;
#endif
    }
  };

  ISSUE_TILE(0)
  // steady state: tile t+1 always in flight while computing tile t
  for (int t = 0; t < nk - 1; ++t) {
    ISSUE_TILE(t + 1)
    WAIT_NEXT();                                     // tile t has landed
    __syncthreads();                                 // ... in every wave
    compute_tile(t);
    __syncthreads();                                 // buf t&1 free for t+2
  }
  // peeled last tile
  WAIT_ALL();
  __syncthreads();
  compute_tile(nk - 1);

  // --- epilogue: bias + relu + bf16 store (C/D layout: VGPR r -> M=r / r+8) ---
  int nb = n0 + wn * 32 + (lane & 15);
  int mb = m0 + wm * 16 + ((lane < 16) ? 0 : 8);
  float b0 = bias[nb];
  float b1v = bias[nb + 16];
#pragma unroll
  for (int r = 0; r < 8; ++r) {
    int m = mb + r;
    float v0 = fmaxf(acc0[r] + b0, 0.0f);
    float v1 = fmaxf(acc1[r] + b1v, 0.0f);
    Cout[(long)m * Nn + nb]      = (__bf16)v0;
    Cout[(long)m * Nn + nb + 16] = (__bf16)v1;
  }
  #undef ISSUE_TILE
  #undef WAIT_NEXT
  #undef WAIT_ALL
}

// ---------------------------------------------------------------------------
// Kernel 6: logits = h2 @ W3 + b3   (one wave32 per edge row)
// ---------------------------------------------------------------------------
__global__ void __launch_bounds__(256)
k_final_dot(const __bf16* __restrict__ h2, const float* __restrict__ W3,
            const float* __restrict__ b3, float* __restrict__ out) {
  int lane = threadIdx.x & 31;
  int e = blockIdx.x * (blockDim.x >> 5) + (threadIdx.x >> 5);
  if (e >= E_EDGES) return;
  float sum = 0.0f;
#pragma unroll
  for (int j = 0; j < 8; ++j) {
    int c = lane * 8 + j;
    sum += (float)h2[(long)e * HID + c] * W3[c];
  }
  for (int off = 16; off > 0; off >>= 1) sum += __shfl_down(sum, off);
  if (lane == 0) out[e] = sum + b3[0];
}

// ---------------------------------------------------------------------------
// Launch
// ---------------------------------------------------------------------------
extern "C" void kernel_launch(void* const* d_in, const int* in_sizes, int n_in,
                              void* d_out, int out_size, void* d_ws, size_t ws_size,
                              hipStream_t stream) {
  const float*         x     = (const float*)d_in[0];
  const unsigned char* adj   = (const unsigned char*)d_in[1];   // bool bytes
  const int*           edges = (const int*)d_in[2];             // [2, E]
  const float*         Ws    = (const float*)d_in[3];
  const float*         bs    = (const float*)d_in[4];
  const float*         gamma = (const float*)d_in[5];
  const float*         beta  = (const float*)d_in[6];
  const float*         W1    = (const float*)d_in[7];
  const float*         b1    = (const float*)d_in[8];
  const float*         W2    = (const float*)d_in[9];
  const float*         b2    = (const float*)d_in[10];
  const float*         W3    = (const float*)d_in[11];
  const float*         b3    = (const float*)d_in[12];
  float* out = (float*)d_out;

  // workspace layout (bytes)
  char* ws = (char*)d_ws;
  unsigned int* Ap   = (unsigned int*)(ws + 0);                 //  2 MB
  unsigned int* A2p  = (unsigned int*)(ws + (2u << 20));        //  2 MB
  __bf16*       feat = (__bf16*)(ws + (4u << 20));              //  8192*288*2
  __bf16*       W1t  = (__bf16*)(ws + 8912896);                 //  256*288*2
  __bf16*       W2t  = (__bf16*)(ws + 9060352);                 //  256*256*2
  __bf16*       h1   = (__bf16*)(ws + 9191424);                 //  8192*256*2
  __bf16*       h2   = (__bf16*)(ws + 13385728);                //  8192*256*2

  // 1) bit-pack adjacency: 4096 rows x 128 words
  k_pack_bits<<<(N_NODES * NWORDS) / 256, 256, 0, stream>>>(adj, Ap);
  // 2) strict two-hop bitsets
  k_twohop<<<N_NODES, 128, 0, stream>>>(Ap, A2p);
  // 3) per-edge DE counts + struct encode + fused feature build
  k_edge_feat<<<E_EDGES / 8, 256, 0, stream>>>(Ap, A2p, edges, x, Ws, bs,
                                               gamma, beta, feat);
  // 4) weight conversion + transpose to bf16 [n][k]
  k_cvt_weights<<<(HID * KPAD + HID * HID) / 256, 256, 0, stream>>>(W1, W2,
                                                                    W1t, W2t);
  // 5) MLP layer 1: [8192,288] @ W1 -> relu -> bf16
  dim3 g1(HID / 64, E_EDGES / 64);
  k_gemm_relu_wmma<<<g1, 256, 0, stream>>>(feat, W1t, b1, h1,
                                           E_EDGES, HID, KPAD);
  // 6) MLP layer 2: [8192,256] @ W2 -> relu -> bf16
  k_gemm_relu_wmma<<<g1, 256, 0, stream>>>(h1, W2t, b2, h2,
                                           E_EDGES, HID, HID);
  // 7) final dot + bias -> f32 logits [8192,1]
  k_final_dot<<<E_EDGES / 8, 256, 0, stream>>>(h2, W3, b3, out);
}
